// CausalSelfAttention_6846177870257
// MI455X (gfx1250) — matrix operand order repaired and checked by reading, not compiled, not run on hardware
//
#include <hip/hip_runtime.h>
#include <math.h>

// ---------------------------------------------------------------------------
// CDNA5 (gfx1250) causal self-attention, bf16 WMMA + async-to-LDS + TDM.
// Pipeline: cvt(f32->bf16, weights pre-transposed) -> QKV GEMM (double-
// buffered LDS via GLOBAL_LOAD_ASYNC_TO_LDS_B128) -> flash attention (K/V
// tiles via TENSOR_LOAD_TO_LDS descriptors) -> proj GEMM.
// ---------------------------------------------------------------------------

typedef __attribute__((ext_vector_type(16))) __bf16 v16bf;
typedef __attribute__((ext_vector_type(8)))  float  v8f;
typedef int          v4i_vs  __attribute__((vector_size(16)));
typedef unsigned int u32x4   __attribute__((vector_size(16)));
typedef int          i32x4   __attribute__((vector_size(16)));
typedef int          i32x8   __attribute__((vector_size(32)));

#if defined(__AMDGCN__) && __has_builtin(__builtin_amdgcn_global_load_async_to_lds_b128)
#define HAVE_ASYNC_LDS 1
#else
#define HAVE_ASYNC_LDS 0
#endif

#if defined(__AMDGCN__) && __has_builtin(__builtin_amdgcn_tensor_load_to_lds) && \
    __has_builtin(__builtin_amdgcn_s_wait_tensorcnt)
#define HAVE_TDM 1
#else
#define HAVE_TDM 0
#endif

union Frag16 { v16bf v; uint4 u[2]; };

__device__ __forceinline__ v8f zero8() {
  v8f z;
#pragma unroll
  for (int i = 0; i < 8; ++i) z[i] = 0.0f;
  return z;
}

__device__ __forceinline__ v16bf load_frag(const __bf16* p0, const __bf16* p1) {
  Frag16 f;
  f.u[0] = *(const uint4*)p0;   // 8 bf16 (16B, aligned)
  f.u[1] = *(const uint4*)p1;
  return f.v;
}

__device__ __forceinline__ v8f wmma_bf16(v16bf a, v16bf b, v8f c) {
  // D(16x16 f32) = A(16x32 bf16) x B(32x16 bf16) + C
  return __builtin_amdgcn_wmma_f32_16x16x32_bf16(
      false, a, false, b, (short)0, c, false, false);
}

// 16B global -> LDS copy: async DMA on CDNA5 (ASYNCcnt), plain copy otherwise.
__device__ __forceinline__ void async_b128(const __bf16* g, __bf16* l) {
#if HAVE_ASYNC_LDS
  __builtin_amdgcn_global_load_async_to_lds_b128(
      (__attribute__((address_space(1))) v4i_vs*)(g),
      (__attribute__((address_space(3))) v4i_vs*)(l), 0, 0);
#else
  *(uint4*)l = *(const uint4*)g;
#endif
}

__device__ __forceinline__ void wait_async() {
#if HAVE_ASYNC_LDS
  asm volatile("s_wait_asynccnt 0x0" ::: "memory");
#endif
}

#if HAVE_TDM
// LDS byte offset of a __shared__ object (ptrtoint of its AS3 address).
__device__ __forceinline__ unsigned lds_off(const void* p) {
  return (unsigned)(unsigned long long)
      (__attribute__((address_space(3))) const void*)p;
}

// Issue one 2D TDM tile load: tile_d1 rows of tile_d0 elements (2B each),
// source rows at stride d0_stride elements; optional LDS row padding.
__device__ __forceinline__ void tdm_load_2d(const void* gaddr, unsigned lds_byte,
                                            unsigned tile_d0, unsigned tile_d1,
                                            unsigned long long d0_stride,
                                            unsigned pad_en, unsigned pad_iv,
                                            unsigned pad_amt) {
  unsigned long long ga = (unsigned long long)gaddr;
  u32x4 g0;
  g0[0] = 1u;                                   // count=1, user mode
  g0[1] = lds_byte;                             // lds_addr (bytes)
  g0[2] = (unsigned)(ga & 0xFFFFFFFFu);         // global_addr[31:0]
  g0[3] = (unsigned)((ga >> 32) & 0x01FFFFFFu)  // global_addr[56:32]
          | (2u << 30);                         // type=2 ("image")
  i32x8 g1;
  g1[0] = (int)((1u << 16) |                    // data_size=1 -> 2 bytes
                (pad_en << 20) | (pad_iv << 22) | (pad_amt << 25));
  g1[1] = (int)((tile_d0 & 0xFFFFu) << 16);     // tensor_dim0[15:0] (=tile_d0)
  g1[2] = (int)(((tile_d0 >> 16) & 0xFFFFu) |   // tensor_dim0[31:16]
                ((tile_d1 & 0xFFFFu) << 16));   // tensor_dim1[15:0]
  g1[3] = (int)(((tile_d1 >> 16) & 0xFFFFu) |   // tensor_dim1[31:16]
                ((tile_d0 & 0xFFFFu) << 16));   // tile_dim0
  g1[4] = (int)(tile_d1 & 0xFFFFu);             // tile_dim1; tile_dim2=0
  g1[5] = (int)(unsigned)(d0_stride & 0xFFFFFFFFu);  // dim0_stride[31:0]
  g1[6] = (int)(unsigned)((d0_stride >> 32) & 0xFFFFu);  // dim0_stride[47:32]
  g1[7] = 0;
  i32x4 z4 = {0, 0, 0, 0};
  i32x8 z8 = {0, 0, 0, 0, 0, 0, 0, 0};
  // amdgpu-toolchain (clang-23) 6-arg form: (g0, g1, g2, g3, g4, cpol)
  __builtin_amdgcn_tensor_load_to_lds(g0, g1, z4, z4, z8, 0);
}
#endif

// ---------------------------------------------------------------------------
// Stage 0: fp32 -> bf16 conversion (optionally transposing weights)
// ---------------------------------------------------------------------------
__global__ void cvt_f32_bf16(const float* __restrict__ src,
                             __bf16* __restrict__ dst, int n) {
  int i = blockIdx.x * blockDim.x + threadIdx.x;
  int stride = gridDim.x * blockDim.x;
  for (; i < n; i += stride) dst[i] = (__bf16)src[i];
}

// src [K x N] row-major -> dst [N x K] bf16 (so GEMM B-tiles are contiguous)
__global__ void cvt_transpose_f32_bf16(const float* __restrict__ src,
                                       __bf16* __restrict__ dst, int K2, int N) {
  int i = blockIdx.x * blockDim.x + threadIdx.x;
  int stride = gridDim.x * blockDim.x;
  int total = K2 * N;
  for (; i < total; i += stride) {
    int kk = i / N, n = i - kk * N;
    dst[(size_t)n * K2 + kk] = (__bf16)src[i];
  }
}

// ---------------------------------------------------------------------------
// Tiled GEMM: C[8192 x N] = A[8192 x 1024] * BwT^T + bias, bf16 in, f32 acc.
// Workgroup tile 128x128, 8 waves x (32x64), LDS double-buffered async loads.
// MODE 0 (N=3072): scatter bf16 into q/k [B,H,T,64] and v [B,H,64,T].
// MODE 1 (N=1024): f32 output.
// ---------------------------------------------------------------------------
template <int N, int MODE>
__global__ __launch_bounds__(256) void gemm_kernel(
    const __bf16* __restrict__ A,    // [8192 x 1024]
    const __bf16* __restrict__ BwT,  // [N x 1024] pre-transposed
    const float* __restrict__ bias,
    __bf16* __restrict__ qout, __bf16* __restrict__ kout,
    __bf16* __restrict__ vout, float* __restrict__ fout) {
  constexpr int K = 1024;
  __shared__ __attribute__((aligned(16))) __bf16 Alds[2][128 * 32];  // [m][k]
  __shared__ __attribute__((aligned(16))) __bf16 Bt[2][128 * 40];    // [n][k] pad

  const int t = threadIdx.x;
  const int lane = t & 31, wid = t >> 5;
  const int wm = wid & 3, wn = wid >> 2;     // 4 waves on M, 2 on N
  const int la = lane & 15, lh = lane >> 4;  // lane column / half
  const int n0 = blockIdx.x * 128;
  const int m0 = blockIdx.y * 128;

  // Issue one 128x32 A slab + 128x32 B slab into LDS buffer `buf` (async).
  auto stage = [&](int k0, int buf) {
    int row = t >> 1, col = (t & 1) * 16;  // 256 thr x 16 halves = 4096 elems
    const __bf16* ga = A + (size_t)(m0 + row) * K + k0 + col;
    __bf16* la_ = &Alds[buf][row * 32 + col];
    async_b128(ga, la_);
    async_b128(ga + 8, la_ + 8);
    const __bf16* gb = BwT + (size_t)(n0 + row) * K + k0 + col;
    __bf16* lb_ = &Bt[buf][row * 40 + col];
    async_b128(gb, lb_);
    async_b128(gb + 8, lb_ + 8);
  };

  v8f acc[2][4];
#pragma unroll
  for (int mt = 0; mt < 2; ++mt)
#pragma unroll
    for (int nt = 0; nt < 4; ++nt) acc[mt][nt] = zero8();

  stage(0, 0);
  wait_async();
  __syncthreads();

  for (int k0 = 0; k0 < K; k0 += 32) {
    const int buf = (k0 >> 5) & 1;
    if (k0 + 32 < K) stage(k0 + 32, buf ^ 1);  // overlap DMA with WMMA

    v16bf afr[2];
#pragma unroll
    for (int mt = 0; mt < 2; ++mt) {
      int m = wm * 32 + mt * 16 + la;
      // A 16x32: lane row m=la, K = {kb..kb+7, 16+kb..16+kb+7}, kb=lh*8
      afr[mt] = load_frag(&Alds[buf][m * 32 + lh * 8],
                          &Alds[buf][m * 32 + 16 + lh * 8]);
    }
#pragma unroll
    for (int nt = 0; nt < 4; ++nt) {
      int n = wn * 64 + nt * 16 + la;
      // B 32x16: lane col n=la, K = {kb..kb+7, 16+kb..16+kb+7}, kb=lh*8
      v16bf bfr = load_frag(&Bt[buf][n * 40 + lh * 8],
                            &Bt[buf][n * 40 + 16 + lh * 8]);
#pragma unroll
      for (int mt = 0; mt < 2; ++mt)
        acc[mt][nt] = wmma_bf16(afr[mt], bfr, acc[mt][nt]);
    }
    wait_async();
    __syncthreads();
  }

  // Epilogue. C layout: lane half (lh) holds rows lh*8+i, column = la.
#pragma unroll
  for (int mt = 0; mt < 2; ++mt)
#pragma unroll
    for (int nt = 0; nt < 4; ++nt) {
      int n_g = n0 + wn * 64 + nt * 16 + la;
      float bv = bias[n_g];
#pragma unroll
      for (int i = 0; i < 8; ++i) {
        int m_g = m0 + wm * 32 + mt * 16 + lh * 8 + i;
        float val = acc[mt][nt][i] + bv;
        if (MODE == 0) {
          int sec = n_g >> 10, c = n_g & 1023;
          int h = c >> 6, d = c & 63;
          int bb = m_g >> 11, tt = m_g & 2047;
          if (sec == 0)
            qout[(((size_t)(bb * 16 + h)) * 2048 + tt) * 64 + d] = (__bf16)val;
          else if (sec == 1)
            kout[(((size_t)(bb * 16 + h)) * 2048 + tt) * 64 + d] = (__bf16)val;
          else  // V stored transposed [B,H,64,T] for the attention PV GEMM
            vout[(((size_t)(bb * 16 + h)) * 64 + d) * 2048 + tt] = (__bf16)val;
        } else {
          fout[(size_t)m_g * N + n_g] = val;
        }
      }
    }
}

// ---------------------------------------------------------------------------
// Flash attention: one workgroup per (b*H+h, 128-query tile).
// 8 waves x 16 query rows. Online softmax, causal mask, scale 1/8.
// K tile [128 key][64 d] and V tile [64 d][128 key] DMA'd into LDS by the
// Tensor Data Mover (one descriptor each; V uses TDM row padding to produce
// the 136-half LDS stride). Falls back to async/plain copies.
// ---------------------------------------------------------------------------
__global__ __launch_bounds__(256) void attn_kernel(
    const __bf16* __restrict__ q, const __bf16* __restrict__ k,
    const __bf16* __restrict__ vT, __bf16* __restrict__ y) {
  constexpr int T = 2048, HD = 64;
  __shared__ __attribute__((aligned(16))) __bf16 Klds[128 * 64];  // [key][d]
  __shared__ __attribute__((aligned(16))) __bf16 Vt[64 * 136];    // [d][key] pad
  __shared__ __attribute__((aligned(16))) __bf16 Plds[8 * 16 * 32];

  const int t = threadIdx.x;
  const int lane = t & 31, wid = t >> 5;
  const int la = lane & 15, lh = lane >> 4;
  const int bh = blockIdx.x;  // b*16 + h
  const int qt = blockIdx.y;  // query tile 0..15
  const int b = bh >> 4, h = bh & 15;
  const size_t head_off = (size_t)bh * T * HD;
  const size_t vrow0 = (size_t)bh * HD * T;  // vT head base
  const int qrow = qt * 128 + wid * 16;

  // Q fragments in registers: 16 rows x 64 d, two K=32 chunks
  const __bf16* qp = q + head_off + (size_t)(qrow + la) * HD + lh * 8;
  v16bf qf[2];
  qf[0] = load_frag(qp + 0, qp + 16);
  qf[1] = load_frag(qp + 32, qp + 48);

  v8f o[4];
#pragma unroll
  for (int nt = 0; nt < 4; ++nt) o[nt] = zero8();
  float rmax[8], rsum[8];
#pragma unroll
  for (int i = 0; i < 8; ++i) { rmax[i] = -INFINITY; rsum[i] = 0.0f; }

  __bf16* Pw = Plds + wid * (16 * 32);
#if HAVE_TDM
  const unsigned ldsK = lds_off(Klds), ldsV = lds_off(Vt);
#endif

  for (int kt = 0; kt <= qt; ++kt) {
    const size_t kv0 = head_off + (size_t)kt * 128 * HD;
#if HAVE_TDM
    if (t == 0) {
      // K tile: 128 rows x 64 halves, contiguous (stride 64), no padding.
      tdm_load_2d(k + kv0, ldsK, 64, 128, 64, 0, 0, 0);
      // V tile: 64 rows x 128 halves at stride T; pad 4 DWORDs per 64 DWORDs
      // so LDS rows land at the 136-half (16B-aligned) stride.
      tdm_load_2d(vT + vrow0 + (size_t)kt * 128, ldsV, 128, 64, T, 1, 5, 3);
      __builtin_amdgcn_s_wait_tensorcnt(0);
    }
#else
#pragma unroll
    for (int c = t; c < 1024; c += 256) {  // K tile, row-major
      int row = c >> 3, col = (c & 7) * 8;
      async_b128(k + kv0 + row * 64 + col, Klds + row * 64 + col);
    }
#pragma unroll
    for (int c = t; c < 1024; c += 256) {  // V tile (already d-major in HBM)
      int row = c >> 4, col = (c & 15) * 8;
      async_b128(vT + vrow0 + (size_t)row * T + kt * 128 + col,
                 Vt + row * 136 + col);
    }
    wait_async();
#endif
    __syncthreads();
    if (kt < qt)  // speculative prefetch of next key tile
      __builtin_prefetch(k + kv0 + 128 * HD + (size_t)t * 32, 0, 1);

    // S = (Q K^T) / 8, causal masked. 8 key sub-tiles of 16.
    v8f s[8];
#pragma unroll
    for (int nt = 0; nt < 8; ++nt) {
      v8f a = zero8();
#pragma unroll
      for (int c = 0; c < 2; ++c) {
        const __bf16* kp = Klds + (nt * 16 + la) * 64 + c * 32 + lh * 8;
        a = wmma_bf16(qf[c], load_frag(kp, kp + 16), a);
      }
      int kg = kt * 128 + nt * 16 + la;
#pragma unroll
      for (int i = 0; i < 8; ++i) {
        int qg = qt * 128 + wid * 16 + lh * 8 + i;
        float x = a[i] * 0.125f;
        s[nt][i] = (kg <= qg) ? x : -INFINITY;
      }
    }

    // Online softmax; row r = lh*8+i lives in 16 lanes of one half-wave.
    float nm[8], alpha[8];
#pragma unroll
    for (int i = 0; i < 8; ++i) {
      float m = rmax[i];
#pragma unroll
      for (int nt = 0; nt < 8; ++nt) m = fmaxf(m, s[nt][i]);
      m = fmaxf(m, __shfl_xor(m, 1, 32));
      m = fmaxf(m, __shfl_xor(m, 2, 32));
      m = fmaxf(m, __shfl_xor(m, 4, 32));
      m = fmaxf(m, __shfl_xor(m, 8, 32));
      nm[i] = m;
      alpha[i] = __expf(rmax[i] - m);
      rmax[i] = m;
    }
#pragma unroll
    for (int nt = 0; nt < 4; ++nt)
#pragma unroll
      for (int i = 0; i < 8; ++i) o[nt][i] *= alpha[i];
#pragma unroll
    for (int i = 0; i < 8; ++i) {
      float sum = 0.0f;
#pragma unroll
      for (int nt = 0; nt < 8; ++nt) {
        float p = __expf(s[nt][i] - nm[i]);
        s[nt][i] = p;
        sum += p;
      }
      sum += __shfl_xor(sum, 1, 32);
      sum += __shfl_xor(sum, 2, 32);
      sum += __shfl_xor(sum, 4, 32);
      sum += __shfl_xor(sum, 8, 32);
      rsum[i] = rsum[i] * alpha[i] + sum;
    }

    // O += P @ V, 32 keys per chunk; P reshaped C-layout -> A-layout via LDS.
#pragma unroll
    for (int c = 0; c < 4; ++c) {
#pragma unroll
      for (int j = 0; j < 2; ++j) {
        int nt = 2 * c + j;
#pragma unroll
        for (int i = 0; i < 8; ++i)
          Pw[(lh * 8 + i) * 32 + j * 16 + la] = (__bf16)s[nt][i];
      }
      asm volatile("s_wait_dscnt 0x0" ::: "memory");  // per-wave LDS RAW fence
      const __bf16* pp = Pw + la * 32 + lh * 8;
      v16bf pf = load_frag(pp, pp + 16);
#pragma unroll
      for (int nt = 0; nt < 4; ++nt) {
        const __bf16* vp = Vt + (nt * 16 + la) * 136 + c * 32 + lh * 8;
        o[nt] = wmma_bf16(pf, load_frag(vp, vp + 16), o[nt]);
      }
    }
    __syncthreads();
  }

  // Normalize and store y[b*T+row][h*64 + d] as bf16.
#pragma unroll
  for (int nt = 0; nt < 4; ++nt)
#pragma unroll
    for (int i = 0; i < 8; ++i) {
      int row = qrow + lh * 8 + i;
      float val = o[nt][i] / rsum[i];
      y[(size_t)(b * T + row) * 1024 + h * 64 + nt * 16 + la] = (__bf16)val;
    }
}

// ---------------------------------------------------------------------------
// Host launcher
// ---------------------------------------------------------------------------
extern "C" void kernel_launch(void* const* d_in, const int* in_sizes, int n_in,
                              void* d_out, int out_size, void* d_ws,
                              size_t ws_size, hipStream_t stream) {
  (void)in_sizes; (void)n_in; (void)out_size; (void)ws_size;
  const float* x  = (const float*)d_in[0];
  const float* Wa = (const float*)d_in[1];
  const float* ba = (const float*)d_in[2];
  const float* Wp = (const float*)d_in[3];
  const float* bp = (const float*)d_in[4];
  float* out = (float*)d_out;

  char* ws = (char*)d_ws;  // ~88 MB used
  __bf16* xb   = (__bf16*)(ws + 0);         // 8192x1024      16 MB
  __bf16* wabT = (__bf16*)(ws + 16777216);  // [3072 x 1024]   6 MB
  __bf16* wpbT = (__bf16*)(ws + 23068672);  // [1024 x 1024]   2 MB
  __bf16* qb   = (__bf16*)(ws + 25165824);  // [B,H,T,64]     16 MB
  __bf16* kb   = (__bf16*)(ws + 41943040);  // [B,H,T,64]     16 MB
  __bf16* vbT  = (__bf16*)(ws + 58720256);  // [B,H,64,T]     16 MB
  __bf16* yb   = (__bf16*)(ws + 75497472);  // 8192x1024      16 MB

  cvt_f32_bf16<<<2048, 256, 0, stream>>>(x, xb, 8192 * 1024);
  cvt_transpose_f32_bf16<<<2048, 256, 0, stream>>>(Wa, wabT, 1024, 3072);
  cvt_transpose_f32_bf16<<<1024, 256, 0, stream>>>(Wp, wpbT, 1024, 1024);

  gemm_kernel<3072, 0><<<dim3(24, 64), 256, 0, stream>>>(
      xb, wabT, ba, qb, kb, vbT, nullptr);
  attn_kernel<<<dim3(64, 16), 256, 0, stream>>>(qb, kb, vbT, yb);
  gemm_kernel<1024, 1><<<dim3(8, 64), 256, 0, stream>>>(
      yb, wpbT, bp, nullptr, nullptr, nullptr, out);
}
